// BiLSTM_Tok_83837761618147
// MI455X (gfx1250) — compile-verified, matrix operand order checked by reading, not run on hardware
//
#include <hip/hip_runtime.h>
#include <math.h>

// ---------------- problem constants ----------------
constexpr int T      = 131072;
constexpr int D      = 256;    // input dim
constexpr int H      = 128;    // per-direction hidden
constexpr int G4     = 512;    // 4*H gate width
constexpr int HID    = 256;    // bidirectional width
constexpr int TAGS   = 10;
constexpr int NSPLIT = 1023;
constexpr int NSEG   = NSPLIT + 1;

// ---------------- workspace layout (bytes) ----------------
constexpr size_t SZ_WIH  = (size_t)G4 * D * 2;          // 512x256 bf16
constexpr size_t SZ_WOM  = (size_t)HID * HID * 2;       // 256x256 bf16 (transposed)
constexpr size_t SZ_PRE  = (size_t)T * G4 * 4;          // 256 MB each
constexpr size_t SZ_HS   = (size_t)T * H * 4;           // 64 MB each

constexpr size_t OFF_WIHF = 0;
constexpr size_t OFF_WIHB = OFF_WIHF + SZ_WIH;
constexpr size_t OFF_WOMT = OFF_WIHB + SZ_WIH;
constexpr size_t OFF_PREF = OFF_WOMT + SZ_WOM;
constexpr size_t OFF_PREB = OFF_PREF + SZ_PRE;
constexpr size_t OFF_HF   = OFF_PREB + SZ_PRE;
constexpr size_t OFF_HB   = OFF_HF + SZ_HS;
constexpr size_t OFF_ATT  = OFF_HB + SZ_HS;

// ---------------- WMMA types / helpers ----------------
typedef __attribute__((ext_vector_type(16))) __bf16 v16bf;
typedef __attribute__((ext_vector_type(8)))  float  v8f;

__device__ __forceinline__ v8f wmma_bf16(v16bf a, v16bf b, v8f c) {
  // D = A(16x32 bf16) * B(32x16 bf16) + C(16x16 f32)
  return __builtin_amdgcn_wmma_f32_16x16x32_bf16(
      /*neg_a=*/false, a, /*neg_b=*/false, b,
      /*c_mod=*/(short)0, c, /*reuse_a=*/false, /*reuse_b=*/false);
}

// ---- CDNA5 async global->LDS copy (ASYNCcnt path, ISA 10.x / 15.18.3) ----
// 16 bytes per lane, copied from global address gptr into LDS at lds_off
__device__ __forceinline__ void async_copy_b128(unsigned lds_off, const void* gptr) {
  asm volatile("global_load_async_to_lds_b128 %0, %1, off"
               :: "v"(lds_off), "v"(gptr) : "memory");
}
__device__ __forceinline__ void wait_async0() {
  asm volatile("s_wait_asynccnt 0x0" ::: "memory");
}
// generic (flat) pointer to __shared__ carries the LDS aperture in the high
// 32 bits; the low 32 bits are the LDS byte address (ISA 10.2)
__device__ __forceinline__ unsigned lds_addr(const void* p) {
  return (unsigned)(size_t)p;
}

// ISA 7.12.2: 16-bit A/B fragment, element e of lane-half `half`:
//   K(e) = (e/8)*16 + half*8 + (e%8)   on row/col = lane%16
#define FRAG_K(e, half) ((((e) >> 3) << 4) + ((half) << 3) + ((e) & 7))

__device__ __forceinline__ v16bf frag_from_bf16(const __bf16* __restrict__ src, int half) {
  v16bf r;
#pragma unroll
  for (int e = 0; e < 16; ++e) r[e] = src[FRAG_K(e, half)];
  return r;
}

__device__ __forceinline__ float sigf(float x) { return 1.0f / (1.0f + __expf(-x)); }

// ---------------- kernel 0: weight precompression ----------------
__global__ __launch_bounds__(256) void prep_weights(
    const float* __restrict__ wihf, const float* __restrict__ wihb,
    const float* __restrict__ wom,
    __bf16* __restrict__ wihf_b, __bf16* __restrict__ wihb_b,
    __bf16* __restrict__ womT_b) {
  int i = blockIdx.x * 256 + threadIdx.x;
  if (i < G4 * D) {
    wihf_b[i] = (__bf16)wihf[i];
    wihb_b[i] = (__bf16)wihb[i];
  }
  if (i < HID * HID) {
    int k = i >> 8, j = i & 255;                // w_omega is [k][j]
    womT_b[(size_t)j * HID + k] = (__bf16)wom[i];
  }
}

// ---------------- kernel 1: pre-activation GEMM (WMMA bf16) ----------------
// pre[dir][s][:] = x[tok(s)] @ W^T + (b_ih + b_hh);  tok = s (fwd) or T-1-s (bwd)
// grid: (T/16, 2), block: 256 (8 waves); wave w owns output cols [w*64, w*64+64)
// A tile (16x256 f32, 16KB) staged once per block via async global->LDS.
__global__ __launch_bounds__(256) void pre_gemm(
    const float* __restrict__ x,
    const __bf16* __restrict__ w_f, const __bf16* __restrict__ w_b,
    const float* __restrict__ bihf, const float* __restrict__ bhhf,
    const float* __restrict__ bihb, const float* __restrict__ bhhb,
    float* __restrict__ pre_f, float* __restrict__ pre_b) {
  __shared__ alignas(16) float tileA[16][D];

  const int dir = blockIdx.y;
  const int mt  = blockIdx.x;
  const int tid = threadIdx.x;

  // ---- async stage of the A tile: 1024 chunks x 16B ----
#pragma unroll
  for (int it = 0; it < 4; ++it) {
    int c    = it * 256 + tid;
    int r    = c >> 6;                 // tile row
    int koff = (c & 63) << 2;          // float offset in row
    int tok  = dir ? (T - 1 - (mt * 16 + r)) : (mt * 16 + r);
    async_copy_b128(lds_addr(&tileA[r][koff]), x + (size_t)tok * D + koff);
  }
  wait_async0();
  __syncthreads();

  const int wave = tid >> 5;
  const int lane = tid & 31;
  const int r16  = lane & 15;
  const int half = lane >> 4;

  const __bf16* W   = dir ? w_b  : w_f;
  const float*  bih = dir ? bihb : bihf;
  const float*  bhh = dir ? bhhb : bhhf;
  float*        out = dir ? pre_b : pre_f;

  v8f acc[4] = {};
#pragma unroll 2
  for (int kc = 0; kc < D / 32; ++kc) {            // 8 K-chunks of 32
    v16bf a;
#pragma unroll
    for (int e = 0; e < 16; ++e)
      a[e] = (__bf16)tileA[r16][kc * 32 + FRAG_K(e, half)];
    v16bf bfr[4];                                  // load all B tiles first,
#pragma unroll
    for (int nt = 0; nt < 4; ++nt) {
      int col = wave * 64 + nt * 16 + r16;
      bfr[nt] = frag_from_bf16(W + (size_t)col * D + kc * 32, half);
    }
#pragma unroll
    for (int nt = 0; nt < 4; ++nt)                 // ...then MMA back-to-back
      acc[nt] = wmma_bf16(a, bfr[nt], acc[nt]);
  }
#pragma unroll
  for (int nt = 0; nt < 4; ++nt) {
    int col = wave * 64 + nt * 16 + r16;
    float bias = bih[col] + bhh[col];
#pragma unroll
    for (int r = 0; r < 8; ++r) {
      int srow = mt * 16 + r + 8 * half;
      out[(size_t)srow * G4 + col] = acc[nt][r] + bias;
    }
  }
}

// ---------------- kernel 2: sequential LSTM scan (one block per direction) ----------------
__global__ __launch_bounds__(512) void lstm_scan(
    const float* __restrict__ pre_f, const float* __restrict__ pre_b,
    const float* __restrict__ whh_f, const float* __restrict__ whh_b,
    const float* __restrict__ h0, const float* __restrict__ c0,
    float* __restrict__ hf, float* __restrict__ hb) {
  const int dir = blockIdx.x;
  const int j   = threadIdx.x;                     // 0..511, one gate-column each
  const float* pre  = dir ? pre_b : pre_f;
  const float* whh  = dir ? whh_b : whh_f;
  float*       hout = dir ? hb    : hf;

  __shared__ alignas(16) float h_s[H];
  __shared__ float g_s[G4];

  float w[H];                                      // w_hh row j in registers
#pragma unroll
  for (int k = 0; k < H; ++k) w[k] = whh[(size_t)j * H + k];

  float c = 0.0f;
  if (j < H) { h_s[j] = h0[dir * H + j]; c = c0[dir * H + j]; }
  __syncthreads();

  for (int t = 0; t < T; ++t) {
    if (t + 16 < T) __builtin_prefetch(&pre[(size_t)(t + 16) * G4 + j], 0, 1);
    float a0 = 0.0f, a1 = 0.0f;                    // two chains -> halve dep depth
#pragma unroll
    for (int k = 0; k < H; k += 4) {               // ds_load_b128 per 4 taps
      float4 hv = *reinterpret_cast<const float4*>(&h_s[k]);
      a0 = fmaf(hv.x, w[k],     a0);
      a1 = fmaf(hv.y, w[k + 1], a1);
      a0 = fmaf(hv.z, w[k + 2], a0);
      a1 = fmaf(hv.w, w[k + 3], a1);
    }
    g_s[j] = a0 + a1 + pre[(size_t)t * G4 + j];
    __syncthreads();
    if (j < H) {
      float gi = g_s[j], gf = g_s[H + j], gg = g_s[2 * H + j], go = g_s[3 * H + j];
      c = sigf(gf) * c + sigf(gi) * tanhf(gg);
      float h = sigf(go) * tanhf(c);
      int outt = dir ? (T - 1 - t) : t;
      hout[(size_t)outt * H + j] = h;
      h_s[j] = h;
    }
    __syncthreads();
  }
}

// ---------------- kernel 3: attention logits (WMMA bf16, fused tanh+reduce) ----------------
// att[t] = sum_j tanh( [hf|hb][t] @ w_omega[:,j] ) * u_omega[j]
// [hf|hb] tile (16x256 f32) staged once per block via async global->LDS.
__global__ __launch_bounds__(256) void attn_logits(
    const float* __restrict__ hf, const float* __restrict__ hb,
    const __bf16* __restrict__ womT, const float* __restrict__ u_omega,
    float* __restrict__ att) {
  __shared__ alignas(16) float tileX[16][HID];
  __shared__ float satt[16];

  const int mt  = blockIdx.x;
  const int tid = threadIdx.x;

#pragma unroll
  for (int it = 0; it < 4; ++it) {
    int c  = it * 256 + tid;
    int r  = c >> 6;                    // tile row (token)
    int cc = c & 63;                    // 16B chunk within the 256-float row
    int t_tok = mt * 16 + r;
    const float* src = (cc < 32) ? (hf + (size_t)t_tok * H + (cc << 2))
                                 : (hb + (size_t)t_tok * H + ((cc - 32) << 2));
    async_copy_b128(lds_addr(&tileX[r][cc << 2]), src);
  }
  if (tid < 16) satt[tid] = 0.0f;
  wait_async0();
  __syncthreads();

  const int wave = tid >> 5;
  const int lane = tid & 31;
  const int r16  = lane & 15;
  const int half = lane >> 4;

  v8f acc[2] = {};
#pragma unroll 2
  for (int kc = 0; kc < HID / 32; ++kc) {
    v16bf a;
#pragma unroll
    for (int e = 0; e < 16; ++e)
      a[e] = (__bf16)tileX[r16][kc * 32 + FRAG_K(e, half)];
    v16bf bfr[2];
#pragma unroll
    for (int nt = 0; nt < 2; ++nt) {
      int col = wave * 32 + nt * 16 + r16;
      bfr[nt] = frag_from_bf16(womT + (size_t)col * HID + kc * 32, half);
    }
#pragma unroll
    for (int nt = 0; nt < 2; ++nt)
      acc[nt] = wmma_bf16(a, bfr[nt], acc[nt]);
  }
#pragma unroll
  for (int nt = 0; nt < 2; ++nt) {
    int col = wave * 32 + nt * 16 + r16;
    float uo = u_omega[col];
#pragma unroll
    for (int r = 0; r < 8; ++r)
      atomicAdd(&satt[r + 8 * half], tanhf(acc[nt][r]) * uo);  // ds_add_f32
  }
  __syncthreads();
  if (tid < 16) att[mt * 16 + tid] = satt[tid];
}

// ---------------- kernel 4: per-segment softmax pool + tag head ----------------
__global__ __launch_bounds__(256) void seg_pool_tag(
    const float* __restrict__ hf, const float* __restrict__ hb,
    const float* __restrict__ att, const int* __restrict__ mask,
    const float* __restrict__ w_tag, const float* __restrict__ b_tag,
    float* __restrict__ out) {
  const int s   = blockIdx.x;
  const int tid = threadIdx.x;
  const int start = (s == 0) ? 0 : mask[s - 1];
  const int end   = (s == NSPLIT) ? T : mask[s];

  __shared__ float red[256];
  __shared__ float ctx[HID];
  __shared__ float m_s, z_s;

  float m = -INFINITY;
  for (int t = start + tid; t < end; t += 256) m = fmaxf(m, att[t]);
  red[tid] = m; __syncthreads();
  for (int off = 128; off > 0; off >>= 1) {
    if (tid < off) red[tid] = fmaxf(red[tid], red[tid + off]);
    __syncthreads();
  }
  if (tid == 0) m_s = red[0];
  __syncthreads();
  m = m_s;

  float z = 0.0f;
  for (int t = start + tid; t < end; t += 256) z += __expf(att[t] - m);
  red[tid] = z; __syncthreads();
  for (int off = 128; off > 0; off >>= 1) {
    if (tid < off) red[tid] += red[tid + off];
    __syncthreads();
  }
  if (tid == 0) z_s = red[0];
  __syncthreads();
  z = z_s;

  float cacc = 0.0f;
  if (end > start) {
    float inv = 1.0f / z;
    const float* src = (tid < H) ? (hf + tid) : (hb + (tid - H));
    for (int t = start; t < end; ++t)
      cacc += (__expf(att[t] - m) * inv) * src[(size_t)t * H];
  }
  ctx[tid] = cacc;
  __syncthreads();

  if (tid < TAGS) {
    float o = b_tag[tid];
#pragma unroll 8
    for (int d = 0; d < HID; ++d) o += ctx[d] * w_tag[tid * HID + d];
    out[s * TAGS + tid] = o;
  }
}

// ---------------- host launcher ----------------
extern "C" void kernel_launch(void* const* d_in, const int* in_sizes, int n_in,
                              void* d_out, int out_size, void* d_ws, size_t ws_size,
                              hipStream_t stream) {
  const float* x      = (const float*)d_in[0];
  const float* h0     = (const float*)d_in[1];
  const float* c0     = (const float*)d_in[2];
  const float* wihf   = (const float*)d_in[3];
  const float* whhf   = (const float*)d_in[4];
  const float* bihf   = (const float*)d_in[5];
  const float* bhhf   = (const float*)d_in[6];
  const float* wihb   = (const float*)d_in[7];
  const float* whhb   = (const float*)d_in[8];
  const float* bihb   = (const float*)d_in[9];
  const float* bhhb   = (const float*)d_in[10];
  const float* wom    = (const float*)d_in[11];
  const float* u_om   = (const float*)d_in[12];
  const float* w_tag  = (const float*)d_in[13];
  const float* b_tag  = (const float*)d_in[14];
  const int*   mask   = (const int*)d_in[15];
  float* out = (float*)d_out;

  char* ws = (char*)d_ws;
  __bf16* wihf_b = (__bf16*)(ws + OFF_WIHF);
  __bf16* wihb_b = (__bf16*)(ws + OFF_WIHB);
  __bf16* womT_b = (__bf16*)(ws + OFF_WOMT);
  float*  pre_f  = (float*)(ws + OFF_PREF);
  float*  pre_b  = (float*)(ws + OFF_PREB);
  float*  hf     = (float*)(ws + OFF_HF);
  float*  hb     = (float*)(ws + OFF_HB);
  float*  att    = (float*)(ws + OFF_ATT);

  prep_weights<<<(G4 * D + 255) / 256, 256, 0, stream>>>(
      wihf, wihb, wom, wihf_b, wihb_b, womT_b);

  pre_gemm<<<dim3(T / 16, 2), 256, 0, stream>>>(
      x, wihf_b, wihb_b, bihf, bhhf, bihb, bhhb, pre_f, pre_b);

  lstm_scan<<<2, 512, 0, stream>>>(pre_f, pre_b, whhf, whhb, h0, c0, hf, hb);

  attn_logits<<<T / 16, 256, 0, stream>>>(hf, hb, womT_b, u_om, att);

  seg_pool_tag<<<NSEG, 256, 0, stream>>>(hf, hb, att, mask, w_tag, b_tag, out);
}
